// DupchessNNUE_25116968747458
// MI455X (gfx1250) — matrix operand order repaired and checked by reading, not compiled, not run on hardware
//
#include <hip/hip_runtime.h>
#include <hip/hip_bf16.h>

#define IN_SZ  40960
#define L1_SZ  256
#define BATCH  4096

typedef __attribute__((ext_vector_type(16))) __bf16 v16bf;
typedef __attribute__((ext_vector_type(8)))  float  v8f;

// ---------------------------------------------------------------------------
// Prep: convert W_in (256 x 40960, f32, row-major) to bf16 once. It then lives
// in L2 (20 MB of 192 MB) for the whole GEMM.
// ---------------------------------------------------------------------------
__global__ void nnue_convert_w(const float* __restrict__ W, __bf16* __restrict__ Wb) {
    int i = (blockIdx.x * blockDim.x + threadIdx.x) * 4;
    float4 f = *(const float4*)(W + i);
    Wb[i + 0] = (__bf16)f.x;
    Wb[i + 1] = (__bf16)f.y;
    Wb[i + 2] = (__bf16)f.z;
    Wb[i + 3] = (__bf16)f.w;
}

// ---------------------------------------------------------------------------
// Fused NNUE forward. Each workgroup: 32 batch rows.
//   M-tile = 64 rows (rows 0-31 = white persp., 32-63 = black persp.)
//   N-tile = 256 (all of L1)
//   8 waves: mWave (2) x nWave (4); wave tile = 32M x 64N = 2x4 WMMA tiles.
// K loop streams f32 features from HBM, converts to bf16 in regs, and does
// v_wmma_f32_16x16x32_bf16 against the L2-resident bf16 weights.
// Epilogue does the perspective lerp + clip and the tiny 512->32->32->1 MLP.
// ---------------------------------------------------------------------------
__global__ __launch_bounds__(256, 1) void nnue_fused(
    const float* __restrict__ wfeat, const float* __restrict__ bfeat,
    const float* __restrict__ wtmv,
    const __bf16* __restrict__ Wb,  const float* __restrict__ b_in,
    const float* __restrict__ W1,   const float* __restrict__ b1,
    const float* __restrict__ W2,   const float* __restrict__ b2,
    const float* __restrict__ W3,   const float* __restrict__ b3,
    float* __restrict__ out)
{
    extern __shared__ float smem[];
    float* sAcc = smem;                  // [64][256] accumulators (64 KB)
    float* sH1  = smem + 64 * 256;       // [8][512] per-wave h1 (16 KB)
    float* sH2  = sH1 + 8 * 512;         // [8][32]  per-wave h2 (1 KB)

    const int tid    = threadIdx.x;
    const int lane   = tid & 31;
    const int waveId = tid >> 5;
    const int laneLo = lane & 15;        // row/col within 16
    const int laneHi = lane >> 4;        // K-half selector
    const int mWave  = waveId >> 2;      // 0 = white rows, 1 = black rows
    const int nWave  = waveId & 3;       // which 64-column slab of L1
    const int mBase  = blockIdx.x * 32;  // batch row base

    // A operand source: white waves read w_features, black waves b_features.
    const float* X = (mWave == 0 ? wfeat : bfeat) + (size_t)mBase * IN_SZ;

    // Per-lane A row pointers (lanes 0-15 hold K 0-15, lanes 16-31 hold
    // K 16-31 => +laneHi*16 element offset, matching B).
    const float* aPtr[2];
    aPtr[0] = X + (size_t)(laneLo)      * IN_SZ + laneHi * 16;
    aPtr[1] = X + (size_t)(16 + laneLo) * IN_SZ + laneHi * 16;

    // Per-lane B row pointers (B 32x16, contiguous k order used here, same as A: lane = column, lanes 0-15 hold
    // K 0-15, lanes 16-31 hold K 16-31 => 16 contiguous bf16 per lane).
    const __bf16* bPtr[4];
#pragma unroll
    for (int nt = 0; nt < 4; ++nt) {
        int col = nWave * 64 + nt * 16 + laneLo;
        bPtr[nt] = Wb + (size_t)col * IN_SZ + laneHi * 16;
    }

    v8f acc[2][4] = {};

    for (int k0 = 0; k0 < IN_SZ; k0 += 32) {
        // ---- A fragments: 16 f32 per lane -> bf16 --------------------------
        v16bf a[2];
#pragma unroll
        for (int mt = 0; mt < 2; ++mt) {
            const float4* p = (const float4*)(aPtr[mt] + k0);
            const float4* q = (const float4*)(aPtr[mt] + k0 + 8);
            float4 f0 = p[0], f1 = p[1];
            float4 f2 = q[0], f3 = q[1];
            __builtin_prefetch(aPtr[mt] + k0 + 256, 0, 0);  // global_prefetch_b8
            a[mt][0]  = (__bf16)f0.x; a[mt][1]  = (__bf16)f0.y;
            a[mt][2]  = (__bf16)f0.z; a[mt][3]  = (__bf16)f0.w;
            a[mt][4]  = (__bf16)f1.x; a[mt][5]  = (__bf16)f1.y;
            a[mt][6]  = (__bf16)f1.z; a[mt][7]  = (__bf16)f1.w;
            a[mt][8]  = (__bf16)f2.x; a[mt][9]  = (__bf16)f2.y;
            a[mt][10] = (__bf16)f2.z; a[mt][11] = (__bf16)f2.w;
            a[mt][12] = (__bf16)f3.x; a[mt][13] = (__bf16)f3.y;
            a[mt][14] = (__bf16)f3.z; a[mt][15] = (__bf16)f3.w;
        }
        // ---- B fragments: 32 contiguous bytes per lane (L2-resident) -------
        v16bf b[4];
#pragma unroll
        for (int nt = 0; nt < 4; ++nt)
            b[nt] = *(const v16bf*)(bPtr[nt] + k0);
        // ---- 8 WMMAs -------------------------------------------------------
#pragma unroll
        for (int mt = 0; mt < 2; ++mt)
#pragma unroll
            for (int nt = 0; nt < 4; ++nt)
                acc[mt][nt] = __builtin_amdgcn_wmma_f32_16x16x32_bf16(
                    false, a[mt], false, b[nt],
                    (short)0, acc[mt][nt], false, false);
    }

    // ---- Epilogue 1: bias add, spill accumulators to LDS ------------------
    // C layout: VGPR r, lanes 0-15 -> (M=r, N=lane); lanes 16-31 -> (M=8+r).
#pragma unroll
    for (int nt = 0; nt < 4; ++nt) {
        int col = nWave * 64 + nt * 16 + laneLo;
        float bias = b_in[col];
#pragma unroll
        for (int mt = 0; mt < 2; ++mt) {
            int rowBase = mWave * 32 + mt * 16 + laneHi * 8;
#pragma unroll
            for (int r = 0; r < 8; ++r)
                sAcc[(rowBase + r) * 256 + col] = acc[mt][nt][r] + bias;
        }
    }
    __syncthreads();

    // ---- Epilogue 2: perspective lerp + clip + tiny MLP per batch row -----
    for (int it = 0; it < 4; ++it) {
        int r = waveId + it * 8;             // 0..31 within this WG
        int batchRow = mBase + r;
        float wtm = wtmv[batchRow];
        float* h1 = sH1 + waveId * 512;

        for (int j = lane; j < 256; j += 32) {
            float w  = sAcc[r * 256 + j];          // white[j]
            float bl = sAcc[(32 + r) * 256 + j];   // black[j]
            float v0 = wtm * w  + (1.0f - wtm) * bl;
            float v1 = wtm * bl + (1.0f - wtm) * w;
            h1[j]       = fminf(fmaxf(v0, 0.0f), 1.0f);
            h1[256 + j] = fminf(fmaxf(v1, 0.0f), 1.0f);
        }
        __syncthreads();

        // h2[i] = clip(W1[i,:] . h1 + b1[i]), lane i of this wave
        {
            float a2 = b1[lane];
            const float* wrow = W1 + lane * 512;
#pragma unroll 8
            for (int j = 0; j < 512; ++j)
                a2 = fmaf(wrow[j], h1[j], a2);
            sH2[waveId * 32 + lane] = fminf(fmaxf(a2, 0.0f), 1.0f);
        }
        __syncthreads();

        // h3 + output dot
        {
            float a3 = b2[lane];
            const float* wrow = W2 + lane * 32;
            const float* h2 = sH2 + waveId * 32;
#pragma unroll
            for (int k = 0; k < 32; ++k)
                a3 = fmaf(wrow[k], h2[k], a3);
            float h3 = fminf(fmaxf(a3, 0.0f), 1.0f);
            float v = h3 * W3[lane];
#pragma unroll
            for (int off = 16; off > 0; off >>= 1)
                v += __shfl_xor(v, off, 32);
            if (lane == 0) out[batchRow] = v + b3[0];
        }
        __syncthreads();
    }
}

// ---------------------------------------------------------------------------
extern "C" void kernel_launch(void* const* d_in, const int* in_sizes, int n_in,
                              void* d_out, int out_size, void* d_ws, size_t ws_size,
                              hipStream_t stream) {
    const float* wfeat = (const float*)d_in[0];
    const float* bfeat = (const float*)d_in[1];
    const float* wtm   = (const float*)d_in[2];
    const float* W_in  = (const float*)d_in[3];
    const float* b_in  = (const float*)d_in[4];
    const float* W1    = (const float*)d_in[5];
    const float* b1    = (const float*)d_in[6];
    const float* W2    = (const float*)d_in[7];
    const float* b2    = (const float*)d_in[8];
    const float* W3    = (const float*)d_in[9];
    const float* b3    = (const float*)d_in[10];
    float* out = (float*)d_out;

    __bf16* Wb = (__bf16*)d_ws;   // 256*40960*2 = 20 MB scratch

    // 1) weight conversion (4 elements/thread)
    const int wElems = L1_SZ * IN_SZ;
    nnue_convert_w<<<wElems / (256 * 4), 256, 0, stream>>>(W_in, Wb);

    // 2) fused GEMM + MLP
    size_t smemBytes = (size_t)(64 * 256 + 8 * 512 + 8 * 32) * sizeof(float); // 83 KB
    hipFuncSetAttribute(reinterpret_cast<const void*>(nnue_fused),
                        hipFuncAttributeMaxDynamicSharedMemorySize, (int)smemBytes);
    nnue_fused<<<BATCH / 32, 256, smemBytes, stream>>>(
        wfeat, bfeat, wtm, Wb, b_in, W1, b1, W2, b2, W3, b3, out);
}
